// DFineMultiheadAttention_81372450390769
// MI455X (gfx1250) — matrix operand order repaired and checked by reading, not compiled, not run on hardware
//
#include <hip/hip_runtime.h>
#include <math.h>

#define B_N   2
#define S_LEN 2048
#define D_DIM 1024
#define H_NUM 16
#define HD_DIM 64

typedef __attribute__((ext_vector_type(16))) __bf16 v16bf;
typedef __attribute__((ext_vector_type(8)))  __bf16 v8bf;
typedef __attribute__((ext_vector_type(8)))  float  v8f;

static __device__ __forceinline__ __bf16 f2bf(float f) {
  union { float f; unsigned u; } x; x.f = f;
  unsigned r = (x.u + 0x7FFFu + ((x.u >> 16) & 1u)) >> 16;   // round-to-nearest-even
  union { unsigned short s; __bf16 b; } y; y.s = (unsigned short)r;
  return y.b;
}

// Async global->LDS copy of 16 bytes (CDNA5 GLOBAL_LOAD_ASYNC_TO_LDS_B128,
// tracked by ASYNCcnt; LDS dest offset in VGPR, 64-bit global addr in VGPR pair).
static __device__ __forceinline__ void async_copy_b128(const void* gptr, void* lptr) {
  unsigned lds = (unsigned)(size_t)lptr;                     // low 32 bits = LDS offset
  unsigned long long ga = (unsigned long long)(size_t)gptr;
  asm volatile("global_load_async_to_lds_b128 %0, %1, off"
               :: "v"(lds), "v"(ga) : "memory");
}
static __device__ __forceinline__ void async_wait_all() {
  asm volatile("s_wait_asynccnt 0x0" ::: "memory");
}

// ---------------------------------------------------------------------------
// Elementwise: hpos = bf16(hs + pos), hsb = bf16(hs)
// ---------------------------------------------------------------------------
__global__ void add_cvt_kernel(const float* __restrict__ hs, const float* __restrict__ pos,
                               __bf16* __restrict__ hpos, __bf16* __restrict__ hsb, int n) {
  int i = blockIdx.x * 256 + threadIdx.x;
  if (i < n) {
    float a = hs[i];
    hpos[i] = f2bf(a + pos[i]);
    hsb[i]  = f2bf(a);
  }
}

// ---------------------------------------------------------------------------
// Weight transpose + bf16 convert:  Wt[n*K + k] = bf16(W[k*N + n])
// ---------------------------------------------------------------------------
__global__ void wT_cvt_kernel(const float* __restrict__ W, __bf16* __restrict__ Wt,
                              int Kd, int Nd) {
  int i = blockIdx.x * 256 + threadIdx.x;
  if (i < Kd * Nd) {
    int k = i / Nd, n = i % Nd;
    Wt[(size_t)n * Kd + k] = f2bf(W[i]);
  }
}

// ---------------------------------------------------------------------------
// bf16 WMMA GEMM:  C[M,N] = (A[M,K] @ Bt[N,K]^T + bias[N]) * scale
// out_mode 0: f32 row-major, 1: bf16 row-major, 2: bf16 V-transposed [B,H,HD,S]
// Block: 256 threads (8 waves), tile 64(M) x 256(N), K-step 32.
// Tiles staged with async global->LDS (no VGPR landing pad), fenced with
// s_wait_asynccnt 0 before the block barrier.
// ---------------------------------------------------------------------------
__global__ __launch_bounds__(256)
void gemm_bf16_wmma(const __bf16* __restrict__ A, const __bf16* __restrict__ Bt,
                    const float* __restrict__ bias, void* __restrict__ Cout,
                    int M, int N, int K, float scale, int out_mode) {
  __shared__ __align__(32) __bf16 lA[64 * 32];
  __shared__ __align__(32) __bf16 lB[256 * 32];

  const int tid    = threadIdx.x;
  const int w      = tid >> 5;
  const int lane   = tid & 31;
  const int ln     = lane & 15;
  const int half   = lane >> 4;
  const int wave_m = w >> 2;        // 0..1
  const int wave_n = w & 3;         // 0..3
  const int m0     = blockIdx.y * 64;
  const int n0     = blockIdx.x * 256;

  v8f acc[2][4];
  for (int i = 0; i < 2; ++i)
    for (int j = 0; j < 4; ++j)
      for (int e = 0; e < 8; ++e) acc[i][j][e] = 0.f;

  const int arow = tid >> 2, ach = tid & 3;

  for (int k0 = 0; k0 < K; k0 += 32) {
    // Async global -> LDS staging (16B per copy, ASYNCcnt-tracked).
    async_copy_b128(&A[(size_t)(m0 + arow) * K + k0 + ach * 8],
                    &lA[arow * 32 + ach * 8]);
    for (int j = 0; j < 4; ++j) {
      int idx = tid + j * 256;
      int brow = idx >> 2, bch = idx & 3;
      async_copy_b128(&Bt[(size_t)(n0 + brow) * K + k0 + bch * 8],
                      &lB[brow * 32 + bch * 8]);
    }
    // Prefetch next K tile into L2 while this one is consumed.
    if (k0 + 32 < K) {
      __builtin_prefetch(&A[(size_t)(m0 + arow) * K + k0 + 32 + ach * 8], 0, 1);
      __builtin_prefetch(&Bt[(size_t)(n0 + tid) * K + k0 + 32], 0, 1);
    }
    async_wait_all();
    __syncthreads();

    // A fragments (16x32: lane = row, K chunk half*16..+15 contiguous)
    v16bf af[2];
    for (int mi = 0; mi < 2; ++mi) {
      int r = wave_m * 32 + mi * 16 + ln;
      v8bf lo = *(const v8bf*)&lA[r * 32 + half * 16];
      v8bf hi = *(const v8bf*)&lA[r * 32 + half * 16 + 8];
      for (int e = 0; e < 8; ++e) { af[mi][e] = lo[e]; af[mi][e + 8] = hi[e]; }
    }
    // B fragments (32x16: lane = column, K chunk half*16..+15 contiguous)
    v16bf bfr[4];
    for (int ni = 0; ni < 4; ++ni) {
      int c = wave_n * 64 + ni * 16 + ln;
      bfr[ni] = *(const v16bf*)&lB[c * 32 + half * 16];
    }
    for (int mi = 0; mi < 2; ++mi)
      for (int ni = 0; ni < 4; ++ni)
        acc[mi][ni] = __builtin_amdgcn_wmma_f32_16x16x32_bf16(
            false, af[mi], false, bfr[ni], (short)0, acc[mi][ni], false, false);
    __syncthreads();
  }

  // Epilogue (C layout: VGPR v -> row v + 8*half, col = ln)
  for (int mi = 0; mi < 2; ++mi)
    for (int ni = 0; ni < 4; ++ni) {
      int col = n0 + wave_n * 64 + ni * 16 + ln;
      float bv = bias ? bias[col] : 0.f;
      for (int v = 0; v < 8; ++v) {
        int row = m0 + wave_m * 32 + mi * 16 + v + half * 8;
        float val = (acc[mi][ni][v] + bv) * scale;
        if (out_mode == 0) {
          ((float*)Cout)[(size_t)row * N + col] = val;
        } else if (out_mode == 1) {
          ((__bf16*)Cout)[(size_t)row * N + col] = f2bf(val);
        } else {  // V transposed: [B, H, HD, S]
          int bb = row >> 11, s = row & (S_LEN - 1);
          int h = col >> 6, e = col & (HD_DIM - 1);
          ((__bf16*)Cout)[(((size_t)(bb * H_NUM + h) * HD_DIM + e) * S_LEN) + s] = f2bf(val);
        }
      }
    }
}

// ---------------------------------------------------------------------------
// Flash attention: per wave one 16-row Q block; S tiles of 32 with online
// softmax in f32; P re-striped C-layout -> A-layout through per-wave LDS.
// Q is pre-scaled by HD^-0.5. V is transposed [B,H,HD,S]. Output bf16 [B*S,D].
// Block: 128 threads (4 waves). Grid: (S/64, H, B).
// ---------------------------------------------------------------------------
__global__ __launch_bounds__(128)
void attn_wmma(const __bf16* __restrict__ Q, const __bf16* __restrict__ Kb,
               const __bf16* __restrict__ Vt, __bf16* __restrict__ Ao) {
  __shared__ __align__(32) __bf16 lP[4][16 * 32];

  const int tid  = threadIdx.x;
  const int w    = tid >> 5;
  const int lane = tid & 31;
  const int ln   = lane & 15;
  const int half = lane >> 4;
  const int t0   = blockIdx.x * 64 + w * 16;
  const int h    = blockIdx.y;
  const int b    = blockIdx.z;

  // Q A-fragments: 16 rows x 64 e  =>  two 16x32 fragments, resident in VGPRs
  const __bf16* qrow = Q + ((size_t)(b * S_LEN + t0 + ln) * D_DIM) + h * HD_DIM;
  v16bf aq[2];
  for (int eh = 0; eh < 2; ++eh) {
    v8bf lo = *(const v8bf*)(qrow + eh * 32 + half * 16);
    v8bf hi = *(const v8bf*)(qrow + eh * 32 + half * 16 + 8);
    for (int e = 0; e < 8; ++e) { aq[eh][e] = lo[e]; aq[eh][e + 8] = hi[e]; }
  }

  float mrow[8], lrow[8];
  v8f accf[4];
  for (int v = 0; v < 8; ++v) { mrow[v] = -3.0e38f; lrow[v] = 0.f; }
  for (int j = 0; j < 4; ++j)
    for (int v = 0; v < 8; ++v) accf[j][v] = 0.f;

  for (int s0 = 0; s0 < S_LEN; s0 += 32) {
    // scores: c0 = cols s0..s0+15, c1 = cols s0+16..s0+31
    v8f c0, c1;
    for (int v = 0; v < 8; ++v) { c0[v] = 0.f; c1[v] = 0.f; }
    for (int eh = 0; eh < 2; ++eh) {
      const __bf16* kp0 = Kb + (size_t)(b * S_LEN + s0 + ln) * D_DIM + h * HD_DIM + eh * 32 + half * 16;
      const __bf16* kp1 = Kb + (size_t)(b * S_LEN + s0 + 16 + ln) * D_DIM + h * HD_DIM + eh * 32 + half * 16;
      v16bf bk0 = *(const v16bf*)kp0;
      v16bf bk1 = *(const v16bf*)kp1;
      c0 = __builtin_amdgcn_wmma_f32_16x16x32_bf16(false, aq[eh], false, bk0, (short)0, c0, false, false);
      c1 = __builtin_amdgcn_wmma_f32_16x16x32_bf16(false, aq[eh], false, bk1, (short)0, c1, false, false);
    }

    // online softmax stats (row = v + 8*half; reduce across the 16-lane half)
    float tm[8];
    for (int v = 0; v < 8; ++v) tm[v] = fmaxf(c0[v], c1[v]);
    for (int off = 8; off >= 1; off >>= 1)
      for (int v = 0; v < 8; ++v) tm[v] = fmaxf(tm[v], __shfl_xor(tm[v], off, 32));

    float alpha[8];
    for (int v = 0; v < 8; ++v) {
      float mn = fmaxf(mrow[v], tm[v]);
      alpha[v] = __expf(mrow[v] - mn);
      mrow[v] = mn;
    }
    v8f p0, p1;
    float rs[8];
    for (int v = 0; v < 8; ++v) {
      p0[v] = __expf(c0[v] - mrow[v]);
      p1[v] = __expf(c1[v] - mrow[v]);
      rs[v] = p0[v] + p1[v];
    }
    for (int off = 8; off >= 1; off >>= 1)
      for (int v = 0; v < 8; ++v) rs[v] += __shfl_xor(rs[v], off, 32);
    for (int v = 0; v < 8; ++v) lrow[v] = lrow[v] * alpha[v] + rs[v];
    for (int j = 0; j < 4; ++j)
      for (int v = 0; v < 8; ++v) accf[j][v] *= alpha[v];

    // P: C-layout f32 -> row-major bf16 in LDS -> A-layout fragment
    for (int v = 0; v < 8; ++v) {
      lP[w][(v + half * 8) * 32 + ln]      = f2bf(p0[v]);
      lP[w][(v + half * 8) * 32 + 16 + ln] = f2bf(p1[v]);
    }
    __syncthreads();
    v16bf pa;
    {
      v8bf lo = *(const v8bf*)&lP[w][ln * 32 + half * 16];
      v8bf hi = *(const v8bf*)&lP[w][ln * 32 + half * 16 + 8];
      for (int e = 0; e < 8; ++e) { pa[e] = lo[e]; pa[e + 8] = hi[e]; }
    }

    // accf += P @ V : V B-fragments (32 s x 16 e), s contiguous in Vt
    for (int j = 0; j < 4; ++j) {
      const __bf16* vp = Vt + ((size_t)((b * H_NUM + h) * HD_DIM) + j * 16 + ln) * S_LEN + s0 + half * 16;
      v16bf bv = *(const v16bf*)vp;
      accf[j] = __builtin_amdgcn_wmma_f32_16x16x32_bf16(false, pa, false, bv, (short)0, accf[j], false, false);
    }
    __syncthreads();
  }

  // normalize and store attn (bf16, row-major [B*S, D])
  for (int j = 0; j < 4; ++j)
    for (int v = 0; v < 8; ++v) {
      size_t row = (size_t)b * S_LEN + t0 + v + half * 8;
      int col = h * HD_DIM + j * 16 + ln;
      Ao[row * D_DIM + col] = f2bf(accf[j][v] / lrow[v]);
    }
}

// ---------------------------------------------------------------------------
extern "C" void kernel_launch(void* const* d_in, const int* in_sizes, int n_in,
                              void* d_out, int out_size, void* d_ws, size_t ws_size,
                              hipStream_t stream) {
  (void)in_sizes; (void)n_in; (void)out_size; (void)ws_size;

  const float* hs  = (const float*)d_in[0];
  const float* pos = (const float*)d_in[1];
  const float* Wq  = (const float*)d_in[2];
  const float* bq  = (const float*)d_in[3];
  const float* Wk  = (const float*)d_in[4];
  const float* bk  = (const float*)d_in[5];
  const float* Wv  = (const float*)d_in[6];
  const float* bv  = (const float*)d_in[7];
  const float* Wo  = (const float*)d_in[8];
  const float* bo  = (const float*)d_in[9];
  float* out = (float*)d_out;

  const size_t MTOK = (size_t)B_N * S_LEN;          // 4096
  const size_t actB = MTOK * D_DIM * sizeof(__bf16); // 8 MB
  const size_t wB   = (size_t)D_DIM * D_DIM * sizeof(__bf16); // 2 MB

  char* ws = (char*)d_ws;
  size_t off = 0;
  auto alloc = [&](size_t bytes) { void* p = ws + off; off += (bytes + 255) & ~(size_t)255; return p; };

  __bf16* hpos = (__bf16*)alloc(actB);
  __bf16* hsb  = (__bf16*)alloc(actB);
  __bf16* qb   = (__bf16*)alloc(actB);
  __bf16* kb   = (__bf16*)alloc(actB);
  __bf16* vT   = (__bf16*)alloc(actB);   // [B, H, HD, S]
  __bf16* WqT  = (__bf16*)alloc(wB);
  __bf16* WkT  = (__bf16*)alloc(wB);
  __bf16* WvT  = (__bf16*)alloc(wB);
  __bf16* WoT  = (__bf16*)alloc(wB);
  __bf16* attn = hpos;                   // hpos dead after Q/K GEMMs -> reuse

  // 1) activation convert
  {
    int n = (int)(MTOK * D_DIM);
    add_cvt_kernel<<<(n + 255) / 256, 256, 0, stream>>>(hs, pos, hpos, hsb, n);
  }
  // 2) weight transpose + convert
  {
    int n = D_DIM * D_DIM;
    int g = (n + 255) / 256;
    wT_cvt_kernel<<<g, 256, 0, stream>>>(Wq, WqT, D_DIM, D_DIM);
    wT_cvt_kernel<<<g, 256, 0, stream>>>(Wk, WkT, D_DIM, D_DIM);
    wT_cvt_kernel<<<g, 256, 0, stream>>>(Wv, WvT, D_DIM, D_DIM);
    wT_cvt_kernel<<<g, 256, 0, stream>>>(Wo, WoT, D_DIM, D_DIM);
  }
  // 3) QKV projections (Q pre-scaled by HD^-0.5)
  {
    dim3 gg(D_DIM / 256, (unsigned)(MTOK / 64));
    gemm_bf16_wmma<<<gg, 256, 0, stream>>>(hpos, WqT, bq, qb, (int)MTOK, D_DIM, D_DIM, 0.125f, 1);
    gemm_bf16_wmma<<<gg, 256, 0, stream>>>(hpos, WkT, bk, kb, (int)MTOK, D_DIM, D_DIM, 1.0f, 1);
    gemm_bf16_wmma<<<gg, 256, 0, stream>>>(hsb,  WvT, bv, vT, (int)MTOK, D_DIM, D_DIM, 1.0f, 2);
  }
  // 4) flash attention
  {
    dim3 ga(S_LEN / 64, H_NUM, B_N);
    attn_wmma<<<ga, 128, 0, stream>>>(qb, kb, vT, attn);
  }
  // 5) output projection -> f32
  {
    dim3 gg(D_DIM / 256, (unsigned)(MTOK / 64));
    gemm_bf16_wmma<<<gg, 256, 0, stream>>>(attn, WoT, bo, out, (int)MTOK, D_DIM, D_DIM, 1.0f, 0);
  }
}
